// FusionCrossAttention_44521630990599
// MI455X (gfx1250) — matrix operand order repaired and checked by reading, not compiled, not run on hardware
//
#include <hip/hip_runtime.h>

// bf16 compute types for CDNA5 WMMA
typedef __bf16 bf16;
typedef __attribute__((ext_vector_type(16))) __bf16 v16bf;
typedef __attribute__((ext_vector_type(8)))  float  v8f;

#define B_   4
#define C_   256
#define N1_  1600   // 40*40
#define HW2_ 6400   // 80*80

// gfx1250 async global->LDS path (guarded; falls back to sync copies)
#if defined(__gfx1250__) && __has_builtin(__builtin_amdgcn_global_load_async_to_lds_b128) && \
    __has_builtin(__builtin_amdgcn_s_wait_asynccnt)
#define USE_ASYNC 1
#else
#define USE_ASYNC 0
#endif

#if USE_ASYNC
typedef int v4i_ __attribute__((ext_vector_type(4)));
#define AS1_ __attribute__((address_space(1)))
#define AS3_ __attribute__((address_space(3)))
// async b128 copy: LDS[l] = MEM[g], tracked by ASYNCcnt
__device__ __forceinline__ void async_cp16(const bf16* g, bf16* l) {
    __builtin_amdgcn_global_load_async_to_lds_b128(
        (AS1_ v4i_*)(AS1_ const void*)g,
        (AS3_ v4i_*)(AS3_ void*)l, 0, 0);
}
#endif

__device__ __forceinline__ float silu_f(float x) { return x / (1.0f + __expf(-x)); }

// ---------------------------------------------------------------------------
// prep kernels (vectorized: 8 elems / thread, 16B stores)
// ---------------------------------------------------------------------------
__global__ void cvt_f32_bf16_x8(const float* __restrict__ src, bf16* __restrict__ dst, int n8) {
    int i = blockIdx.x * 256 + threadIdx.x;
    if (i >= n8) return;
    const float4* s = (const float4*)src + (size_t)i * 2;
    float4 a = s[0], b = s[1];
    bf16 tmp[8] = {(bf16)a.x, (bf16)a.y, (bf16)a.z, (bf16)a.w,
                   (bf16)b.x, (bf16)b.y, (bf16)b.z, (bf16)b.w};
    *(uint4*)(dst + (size_t)i * 8) = *(const uint4*)tmp;
}

// space-to-depth: ysd[b][p*256+c][oh*40+ow] = y[b][c][2*oh+roff][2*ow+coff]
// concat order: (0::2,0::2), (1::2,0::2), (0::2,1::2), (1::2,1::2)
__global__ void s2d_kernel(const float* __restrict__ y, bf16* __restrict__ ysd) {
    int i = blockIdx.x * 256 + threadIdx.x;        // over B*1024*200
    if (i >= B_ * 1024 * (N1_ / 8)) return;
    int n8 = i % (N1_ / 8);
    int t  = i / (N1_ / 8);
    int ch = t % 1024;
    int b  = t / 1024;
    int p = ch >> 8, c = ch & 255;
    int n  = n8 * 8;
    int oh = n / 40, ow = n % 40;                  // ow in {0,8,16,24,32}: row-local
    int ih = 2 * oh + (p & 1);
    int iw = 2 * ow + (p >> 1);
    const float* row = y + ((size_t)b * C_ + c) * HW2_ + ih * 80 + iw;
    bf16 tmp[8];
#pragma unroll
    for (int u = 0; u < 8; ++u) tmp[u] = (bf16)row[2 * u];
    *(uint4*)(ysd + (size_t)i * 8) = *(const uint4*)tmp;
}

// ---------------------------------------------------------------------------
// B-tile loaders + epilogues for the generic WMMA GEMM
// ---------------------------------------------------------------------------
struct BDirect {          // B[k][n] = p[b*bstride + k*ld + n]
    const bf16* p; long bstride; int ld; int N;
    __device__ bf16 load(int b, int k, int n) const {
        return (n < N) ? p[(size_t)b * bstride + (size_t)k * ld + n] : (bf16)0.0f;
    }
    __device__ void load16(int b, int k, int n, bf16* dst) const {
        if (n + 16 <= N) {                     // fast path: 4 x b64 (32B, aligned)
            const uint2* s = (const uint2*)(p + (size_t)b * bstride + (size_t)k * ld + n);
            uint2* d = (uint2*)dst;
#pragma unroll
            for (int u = 0; u < 4; ++u) d[u] = s[u];
        } else {
#pragma unroll
            for (int t = 0; t < 16; ++t) dst[t] = load(b, k, n + t);
        }
    }
};
struct BIm2col {          // 3x3 pad=1 im2col of yc [B][256][1600], k = ic*9+ky*3+kx
    const bf16* yc;
    __device__ bf16 load(int b, int k, int n) const {
        if (n >= N1_) return (bf16)0.0f;
        int ic = k / 9, rem = k - ic * 9;
        int ky = rem / 3, kx = rem - ky * 3;
        int oh = n / 40, ow = n - oh * 40;
        int ih = oh + ky - 1, iw = ow + kx - 1;
        if ((unsigned)ih < 40u && (unsigned)iw < 40u)
            return yc[((size_t)b * C_ + ic) * N1_ + ih * 40 + iw];
        return (bf16)0.0f;
    }
    __device__ void load16(int b, int k, int n, bf16* dst) const {
        int ic = k / 9, rem = k - ic * 9;      // per-thread constants
        int ky = rem / 3, kx = rem - ky * 3;
        int oh = n / 40, ow = n - oh * 40;
        int ih = oh + ky - 1;
        const bf16* base = yc + ((size_t)b * C_ + ic) * N1_;
#pragma unroll
        for (int t = 0; t < 16; ++t) {         // incremental row/col, no div per elem
            int iw = ow + kx - 1;
            bf16 v = (bf16)0.0f;
            if ((unsigned)ih < 40u && (unsigned)iw < 40u && (n + t) < N1_)
                v = base[ih * 40 + iw];
            dst[t] = v;
            if (++ow == 40) { ow = 0; ++ih; }
        }
    }
};

struct EpiCut {           // BN fold, no act -> Yc bf16 [B][256][1600]
    bf16* out; const float* s; const float* bb;
    __device__ void store(int b, int m, int n, float a) const {
        out[((size_t)b * C_ + m) * N1_ + n] = (bf16)(a * s[m] + bb[m]);
    }
};
struct EpiQ {             // BN+SiLU -> Q bf16 [B][8][1600][32]  (n-major rows)
    bf16* out; const float* s; const float* bb;
    __device__ void store(int b, int m, int n, float a) const {
        float v = silu_f(a * s[m] + bb[m]);
        out[(((size_t)b * 8 + (m >> 5)) * N1_ + n) * 32 + (m & 31)] = (bf16)v;
    }
};
struct EpiKV {            // BN+SiLU; K stored TRANSPOSED Kt[B][8][32][1600], V [B][8][1600][32]
    bf16* ktout; bf16* vout; const float* s; const float* bb;
    __device__ void store(int b, int m, int n, float a) const {
        float v = silu_f(a * s[m] + bb[m]);
        if (m < 256) {
            ktout[(((size_t)b * 8 + (m >> 5)) * 32 + (m & 31)) * N1_ + n] = (bf16)v;
        } else {
            int mm = m - 256;
            vout[(((size_t)b * 8 + (mm >> 5)) * N1_ + n) * 32 + (mm & 31)] = (bf16)v;
        }
    }
};
struct EpiProj {          // BN+SiLU -> f32 NCHW output
    float* out; const float* s; const float* bb;
    __device__ void store(int b, int m, int n, float a) const {
        out[((size_t)b * C_ + m) * N1_ + n] = silu_f(a * s[m] + bb[m]);
    }
};

// ---------------------------------------------------------------------------
// Generic 128x128x32 LDS-tiled bf16 WMMA GEMM: Out = epi(A[M][K] x B[K][N])
//   256 threads = 8 waves (2x4); each wave: 64x32 = 4x2 tiles of 16x16.
//   M % 128 == 0, K % 32 == 0 required (true for all uses); N guarded.
//   A tile staged via gfx1250 async global->LDS when available.
// ---------------------------------------------------------------------------
template <class BL, class EP>
__global__ __launch_bounds__(256) void gemm_wmma(const bf16* __restrict__ A,
                                                 BL bl, EP ep, int N, int K) {
    constexpr int TM = 128, TKc = 32, LA = 40, LB = 132;   // LA rows 80B (16B mult)
    __shared__ bf16 As[TM * LA];
    __shared__ bf16 Bs[TKc * LB];
    const int b    = blockIdx.z;
    const int m0   = blockIdx.y * TM;
    const int n0   = blockIdx.x * 128;
    const int tid  = threadIdx.x;
    const int lane = tid & 31, wave = tid >> 5;
    const int wm = wave & 1, wn = wave >> 1;
    const int ml = lane & 15, half = lane >> 4;

    v8f acc[4][2] = {};

    for (int k0 = 0; k0 < K; k0 += TKc) {
        // ---- A-tile: 128x32 bf16, 32B per thread
        {
            int r = tid >> 1;
            int c = (tid & 1) * 16;
            const bf16* g = A + (size_t)(m0 + r) * K + k0 + c;
            bf16* l = As + r * LA + c;
#if USE_ASYNC
            async_cp16(g, l);
            async_cp16(g + 8, l + 8);
#else
            const uint4* s4 = (const uint4*)g;
            uint4* d4 = (uint4*)l;
            d4[0] = s4[0];
            d4[1] = s4[1];
#endif
            if (k0 + TKc < K)  // prefetch next A tile (global_prefetch_b8)
                __builtin_prefetch(g + TKc, 0, 2);
        }
        // ---- B-tile: 32x128 via loader, 16 elems per thread
        {
            int r  = tid >> 3;
            int c0 = (tid & 7) * 16;
            bl.load16(b, k0 + r, n0 + c0, Bs + r * LB + c0);
        }
#if USE_ASYNC
        __builtin_amdgcn_s_wait_asynccnt(0);
#endif
        __syncthreads();

        // ---- fragments per ISA 7.12.2 layouts
        v16bf af[4], bfr[2];
#pragma unroll
        for (int mt = 0; mt < 4; ++mt) {
            const bf16* arow = As + (wm * 64 + mt * 16 + ml) * LA;
#pragma unroll
            for (int j = 0; j < 8; ++j) {
                int kk = (j < 4) ? (2 * j + 8 * half) : (16 + 2 * (j - 4) + 8 * half);
                af[mt][2 * j]     = arow[kk];
                af[mt][2 * j + 1] = arow[kk + 1];
            }
        }
#pragma unroll
        for (int nt = 0; nt < 2; ++nt) {
            const bf16* bcol = Bs + wn * 32 + nt * 16 + ml;
#pragma unroll
            for (int t = 0; t < 16; ++t) {
                int kk = (t < 8) ? (8 * half + t) : (16 + 8 * half + (t - 8));
                bfr[nt][t] = bcol[kk * LB];
            }
        }
#pragma unroll
        for (int mt = 0; mt < 4; ++mt)
#pragma unroll
            for (int nt = 0; nt < 2; ++nt)
                acc[mt][nt] = __builtin_amdgcn_wmma_f32_16x16x32_bf16(
                    false, af[mt], false, bfr[nt], (short)0, acc[mt][nt], false, false);
        __syncthreads();
    }

    // ---- epilogue (C/D layout: lane->col, vgpr->row)
#pragma unroll
    for (int mt = 0; mt < 4; ++mt)
#pragma unroll
        for (int nt = 0; nt < 2; ++nt) {
            int n = n0 + wn * 32 + nt * 16 + ml;
            if (n < N) {
#pragma unroll
                for (int r = 0; r < 8; ++r) {
                    int m = m0 + wm * 64 + mt * 16 + r + 8 * half;
                    ep.store(b, m, n, acc[mt][nt][r]);
                }
            }
        }
}

// ---------------------------------------------------------------------------
// Attention stage 1: M[32][32] = scale * K^T x V  per (b,h).
//   K is pre-transposed (Kt[d][n]) so A-fragments are contiguous bf16 pairs.
//   4 waves split-K over 50 chunks of 32; LDS f32 atomic reduction (ds_add_f32).
// ---------------------------------------------------------------------------
__global__ __launch_bounds__(128) void ktv_kernel(const bf16* __restrict__ Kt,
                                                  const bf16* __restrict__ Vmat,
                                                  bf16* __restrict__ Mout, float scale) {
    __shared__ float Msum[32 * 32];
    const int bh = blockIdx.x;
    const bf16* Kg = Kt   + (size_t)bh * 32 * N1_;   // [d][n]
    const bf16* Vg = Vmat + (size_t)bh * N1_ * 32;   // [n][d]
    const int tid = threadIdx.x, lane = tid & 31, wave = tid >> 5;
    const int ml = lane & 15, half = lane >> 4;

    for (int i = tid; i < 1024; i += 128) Msum[i] = 0.0f;
    __syncthreads();

    v8f acc[2][2] = {};
    for (int ch = wave; ch < 50; ch += 4) {
        int nb = ch * 32;
        v16bf af[2], bfr[2];
#pragma unroll
        for (int mt = 0; mt < 2; ++mt) {
            const bf16* arow = Kg + (size_t)(mt * 16 + ml) * N1_ + nb;  // A[m=d][k=n]
#pragma unroll
            for (int j = 0; j < 8; ++j) {
                int kk = (j < 4) ? (2 * j + 8 * half) : (16 + 2 * (j - 4) + 8 * half);
                af[mt][2 * j]     = arow[kk];
                af[mt][2 * j + 1] = arow[kk + 1];
            }
        }
#pragma unroll
        for (int nt = 0; nt < 2; ++nt) {
            const bf16* col = Vg + (size_t)nb * 32 + nt * 16 + ml;
#pragma unroll
            for (int t = 0; t < 16; ++t) {
                int kk = (t < 8) ? (8 * half + t) : (16 + 8 * half + (t - 8));
                bfr[nt][t] = col[(size_t)kk * 32];
            }
        }
#pragma unroll
        for (int mt = 0; mt < 2; ++mt)
#pragma unroll
            for (int nt = 0; nt < 2; ++nt)
                acc[mt][nt] = __builtin_amdgcn_wmma_f32_16x16x32_bf16(
                    false, af[mt], false, bfr[nt], (short)0, acc[mt][nt], false, false);
    }
    // cross-wave reduce via LDS float atomics (ds_add_f32)
#pragma unroll
    for (int mt = 0; mt < 2; ++mt)
#pragma unroll
        for (int nt = 0; nt < 2; ++nt)
#pragma unroll
            for (int r = 0; r < 8; ++r) {
                int d1 = mt * 16 + r + 8 * half;
                int d2 = nt * 16 + ml;
                atomicAdd(&Msum[d1 * 32 + d2], acc[mt][nt][r]);
            }
    __syncthreads();
    for (int i = tid; i < 1024; i += 128)
        Mout[(size_t)bh * 1024 + i] = (bf16)(Msum[i] * scale);
}

// ---------------------------------------------------------------------------
// Attention stage 2: O[1600][32] = Q[1600][32] x M[32][32]  per (b,h),
// scattered to channel-major Ob [B][256][1600]. K=32 -> 1 WMMA per tile.
// ---------------------------------------------------------------------------
__global__ __launch_bounds__(128) void qm_kernel(const bf16* __restrict__ Q,
                                                 const bf16* __restrict__ Mb,
                                                 bf16* __restrict__ Ob) {
    __shared__ bf16 Ms[32 * 32];
    const int bh = blockIdx.y;
    const int b = bh >> 3, h = bh & 7;
    const int n0 = blockIdx.x * 128;
    const int tid = threadIdx.x, lane = tid & 31, wave = tid >> 5;
    const int ml = lane & 15, half = lane >> 4;
    const bf16* Qg = Q + (size_t)bh * N1_ * 32;

    for (int i = tid; i < 1024; i += 128) Ms[i] = Mb[(size_t)bh * 1024 + i];
    __syncthreads();

    v16bf af[2], bfr[2];
#pragma unroll
    for (int mt = 0; mt < 2; ++mt) {
        int row = n0 + wave * 32 + mt * 16 + ml;
        if (row > N1_ - 1) row = N1_ - 1;   // clamp loads; stores are guarded
#pragma unroll
        for (int j = 0; j < 8; ++j) {
            int kk = (j < 4) ? (2 * j + 8 * half) : (16 + 2 * (j - 4) + 8 * half);
            af[mt][2 * j]     = Qg[(size_t)row * 32 + kk];
            af[mt][2 * j + 1] = Qg[(size_t)row * 32 + kk + 1];
        }
    }
#pragma unroll
    for (int nt = 0; nt < 2; ++nt)
#pragma unroll
        for (int t = 0; t < 16; ++t) {
            int kk = (t < 8) ? (8 * half + t) : (16 + 8 * half + (t - 8));
            bfr[nt][t] = Ms[kk * 32 + nt * 16 + ml];
        }

    v8f acc[2][2] = {};
#pragma unroll
    for (int mt = 0; mt < 2; ++mt)
#pragma unroll
        for (int nt = 0; nt < 2; ++nt)
            acc[mt][nt] = __builtin_amdgcn_wmma_f32_16x16x32_bf16(
                false, af[mt], false, bfr[nt], (short)0, acc[mt][nt], false, false);

#pragma unroll
    for (int mt = 0; mt < 2; ++mt)
#pragma unroll
        for (int nt = 0; nt < 2; ++nt)
#pragma unroll
            for (int r = 0; r < 8; ++r) {
                int pos = n0 + wave * 32 + mt * 16 + r + 8 * half;
                if (pos < N1_) {
                    int d = nt * 16 + ml;
                    Ob[((size_t)b * C_ + h * 32 + d) * N1_ + pos] = (bf16)acc[mt][nt][r];
                }
            }
}

// ---------------------------------------------------------------------------
// host launcher
// ---------------------------------------------------------------------------
extern "C" void kernel_launch(void* const* d_in, const int* in_sizes, int n_in,
                              void* d_out, int out_size, void* d_ws, size_t ws_size,
                              hipStream_t stream) {
    const float* x      = (const float*)d_in[0];
    const float* y      = (const float*)d_in[1];
    const float* w_cut  = (const float*)d_in[2];
    const float* s_cut  = (const float*)d_in[3];
    const float* b_cut  = (const float*)d_in[4];
    const float* w_q    = (const float*)d_in[5];
    const float* s_q    = (const float*)d_in[6];
    const float* b_q    = (const float*)d_in[7];
    const float* w_kv   = (const float*)d_in[8];
    const float* s_kv   = (const float*)d_in[9];
    const float* b_kv   = (const float*)d_in[10];
    const float* w_proj = (const float*)d_in[11];
    const float* s_proj = (const float*)d_in[12];
    const float* b_proj = (const float*)d_in[13];
    (void)in_sizes; (void)n_in; (void)out_size; (void)ws_size;

    // carve bf16 scratch from workspace (~36 MB total), 256B-aligned
    char* ws = (char*)d_ws;
    size_t off = 0;
    auto carve = [&](size_t bytes) -> char* {
        char* p = ws + off;
        off = (off + bytes + 255) & ~(size_t)255;
        return p;
    };
    bf16* Wcut  = (bf16*)carve((size_t)256 * 1024 * 2);
    bf16* Wq    = (bf16*)carve((size_t)256 * 256 * 2);
    bf16* Wkv   = (bf16*)carve((size_t)512 * 2304 * 2);
    bf16* Wproj = (bf16*)carve((size_t)256 * 256 * 2);
    bf16* Xb    = (bf16*)carve((size_t)B_ * C_ * N1_ * 2);
    bf16* Ysd   = (bf16*)carve((size_t)B_ * 1024 * N1_ * 2);
    bf16* Yc    = (bf16*)carve((size_t)B_ * C_ * N1_ * 2);
    bf16* Qb    = (bf16*)carve((size_t)B_ * C_ * N1_ * 2);
    bf16* Ktb   = (bf16*)carve((size_t)B_ * C_ * N1_ * 2);   // transposed K
    bf16* Vb    = (bf16*)carve((size_t)B_ * C_ * N1_ * 2);
    bf16* Mb    = (bf16*)carve((size_t)B_ * 8 * 32 * 32 * 2);
    bf16* Ob    = (bf16*)carve((size_t)B_ * C_ * N1_ * 2);

    // --- stage 0: conversions / space-to-depth (all sizes divisible by 8)
    auto grid8 = [](int n) { return dim3((n / 8 + 255) / 256); };
    cvt_f32_bf16_x8<<<grid8(256 * 1024), 256, 0, stream>>>(w_cut, Wcut, 256 * 1024 / 8);
    cvt_f32_bf16_x8<<<grid8(256 * 256), 256, 0, stream>>>(w_q, Wq, 256 * 256 / 8);
    cvt_f32_bf16_x8<<<grid8(512 * 2304), 256, 0, stream>>>(w_kv, Wkv, 512 * 2304 / 8);
    cvt_f32_bf16_x8<<<grid8(256 * 256), 256, 0, stream>>>(w_proj, Wproj, 256 * 256 / 8);
    cvt_f32_bf16_x8<<<grid8(B_ * C_ * N1_), 256, 0, stream>>>(x, Xb, B_ * C_ * N1_ / 8);
    s2d_kernel<<<grid8(B_ * 1024 * N1_), 256, 0, stream>>>(y, Ysd);

    const int NB = (N1_ + 127) / 128;   // 13

    // --- cut: Yc = BN(Wcut x Ysd), M=256 K=1024
    gemm_wmma<<<dim3(NB, 2, B_), 256, 0, stream>>>(
        Wcut, BDirect{Ysd, (long)1024 * N1_, N1_, N1_}, EpiCut{Yc, s_cut, b_cut}, N1_, 1024);

    // --- q: Q = SiLU(BN(Wq x Xb)), M=256 K=256
    gemm_wmma<<<dim3(NB, 2, B_), 256, 0, stream>>>(
        Wq, BDirect{Xb, (long)C_ * N1_, N1_, N1_}, EpiQ{Qb, s_q, b_q}, N1_, 256);

    // --- kv: Kt,V = SiLU(BN(Wkv x im2col3x3(Yc))), M=512 K=2304
    gemm_wmma<<<dim3(NB, 4, B_), 256, 0, stream>>>(
        Wkv, BIm2col{Yc}, EpiKV{Ktb, Vb, s_kv, b_kv}, N1_, 2304);

    // --- attention (no softmax): O = Q (scale * K^T V)   [exact reassociation]
    ktv_kernel<<<B_ * 8, 128, 0, stream>>>(Ktb, Vb, Mb, 0.17677669529663687f);
    qm_kernel<<<dim3(NB, B_ * 8), 128, 0, stream>>>(Qb, Mb, Ob);

    // --- proj: out = SiLU(BN(Wproj x Ob)), f32 NCHW
    gemm_wmma<<<dim3(NB, 2, B_), 256, 0, stream>>>(
        Wproj, BDirect{Ob, (long)C_ * N1_, N1_, N1_}, EpiProj{(float*)d_out, s_proj, b_proj},
        N1_, 256);
}
